// Model_70557722739212
// MI455X (gfx1250) — matrix operand order repaired and checked by reading, compiled, not run on hardware
//
#include <hip/hip_runtime.h>
#include <hip/hip_bf16.h>

typedef __attribute__((ext_vector_type(16))) __bf16 v16bf;
typedef __attribute__((ext_vector_type(8)))  float  v8f;

union ABu { v16bf bf; unsigned u[8]; };

#define LB     5           // l-slots per block
#define BCOLS  128         // 5*25 = 125 real cols, padded to 128
#define BROWS  192         // 3 partitions * 64 channels
#define CT     64          // channels per block
#define ROWDW  17          // 16 dwords (32 bf16) per k-row + 1 pad dword
#define WLDS_DW (BROWS*ROWDW)   // 3264 dwords
#define XLDS_DW (BCOLS*ROWDW)   // 2176 dwords

__device__ __forceinline__ unsigned pack_bf16(float a, float b) {
    unsigned ua = __builtin_bit_cast(unsigned, a);
    unsigned ub = __builtin_bit_cast(unsigned, b);
    ua += 0x7FFFu + ((ua >> 16) & 1u);   // RNE
    ub += 0x7FFFu + ((ub >> 16) & 1u);
    return (ua >> 16) | (ub & 0xFFFF0000u);
}

// ---------------------------------------------------------------------------
// Kernel 1: bf16 WMMA GEMM (1x1 conv) + bias + fused spatial aggregation.
// Computes s[n,c,l,w] = sum_p sum_v (W x + b)[n,p,c,l,v] * (A_p*edge_p)[v,w]
// ---------------------------------------------------------------------------
__global__ __launch_bounds__(256)
void stgcn_gemm_spatial(const float* __restrict__ x,
                        const float* __restrict__ A1,
                        const float* __restrict__ A2,
                        const float* __restrict__ A3,
                        const float* __restrict__ conv_w,
                        const float* __restrict__ conv_b,
                        const float* __restrict__ edge,
                        float* __restrict__ s_out)
{
    __shared__ float Ylds[CT * BCOLS];     // 32 KB staging (per partition phase)
    __shared__ float Asc[3 * 25 * 25];     // scaled adjacency
    __shared__ float biasLds[BROWS];
    unsigned* Wlds = (unsigned*)Ylds;                  // aliases Ylds (<=32KB)
    unsigned* Xlds = ((unsigned*)Ylds) + WLDS_DW;

    const int tid  = threadIdx.x;
    const int lane = tid & 31;
    const int wave = tid >> 5;
    const int half = (lane >> 4) & 1;
    const int lr   = lane & 15;

    const int n  = blockIdx.z;
    const int c0 = blockIdx.y * CT;
    const int l0 = blockIdx.x * LB;
    const int nl = (1024 - l0 < LB) ? (1024 - l0) : LB;
    const int ncols = nl * 25;

    // preload scaled adjacency and bias (consumed only after later barriers)
    for (int idx = tid; idx < 3 * 625; idx += 256) {
        int p = idx / 625, vw = idx - p * 625;
        const float* Ap = (p == 0) ? A1 : ((p == 1) ? A2 : A3);
        Asc[idx] = Ap[vw] * edge[p * 625 + vw];
    }
    for (int idx = tid; idx < BROWS; idx += 256)
        biasLds[idx] = conv_b[(idx >> 6) * 256 + c0 + (idx & 63)];

    const float* xn = x + (size_t)n * 256 * 1024 * 25;
    const int jbase = l0 * 25;

    v8f acc[6][2];
    #pragma unroll
    for (int i = 0; i < 6; i++)
        #pragma unroll
        for (int j = 0; j < 2; j++)
            #pragma unroll
            for (int q = 0; q < 8; q++) acc[i][j][q] = 0.0f;

    const int wMb = (wave >> 2) * 96;   // wave M base (rows), 2 strips of 96
    const int wNb = (wave & 3) * 32;    // wave N base (cols), 4 strips of 32

    for (int k0 = 0; k0 < 256; k0 += 32) {
        __syncthreads();
        // W tile: 192 rows x 32 k, fp32 -> packed bf16 pairs
        for (int idx = tid; idx < BROWS * 16; idx += 256) {
            int r = idx >> 4, dw = idx & 15;
            int o = (r >> 6) * 256 + c0 + (r & 63);
            const float* wp = conv_w + (size_t)o * 256 + k0 + 2 * dw;
            Wlds[r * ROWDW + dw] = pack_bf16(wp[0], wp[1]);
        }
        // X tile: 32 k x 128 cols (coalesced over cols), zero-pad tail cols
        for (int idx = tid; idx < 16 * BCOLS; idx += 256) {
            int dw = idx >> 7;           // k-pair index 0..15
            int j  = idx & 127;          // column 0..127
            float v0 = 0.f, v1 = 0.f;
            if (j < ncols) {
                const float* xp = xn + (size_t)(k0 + 2 * dw) * (1024 * 25) + jbase + j;
                v0 = xp[0];
                v1 = xp[1024 * 25];
            }
            Xlds[j * ROWDW + dw] = pack_bf16(v0, v1);
        }
        __syncthreads();

        // B fragments (32x16 bf16): lanes 0-15 K=0..7,16..23, lanes 16-31 K=8..15,24..31
        ABu bfr[2];
        #pragma unroll
        for (int ct = 0; ct < 2; ct++) {
            int col = wNb + ct * 16 + lr;
            #pragma unroll
            for (int i = 0; i < 8; i++)
                bfr[ct].u[i] = Xlds[col * ROWDW + (i & 4) * 2 + 4 * half + (i & 3)];
        }
        // A fragments (16x32 bf16) per ISA layout, then WMMA
        #pragma unroll
        for (int mt = 0; mt < 6; mt++) {
            int m = wMb + mt * 16 + lr;
            ABu af;
            #pragma unroll
            for (int i = 0; i < 4; i++)
                af.u[i] = Wlds[m * ROWDW + 4 * half + i];          // K 0-7 / 8-15
            #pragma unroll
            for (int i = 4; i < 8; i++)
                af.u[i] = Wlds[m * ROWDW + 8 + 4 * half + (i - 4)]; // K 16-23 / 24-31
            #pragma unroll
            for (int ct = 0; ct < 2; ct++)
                acc[mt][ct] = __builtin_amdgcn_wmma_f32_16x16x32_bf16(
                    false, af.bf, false, bfr[ct].bf, (short)0,
                    acc[mt][ct], false, false);
        }
    }

    // Epilogue: per-partition y staging + spatial aggregation, summed over p.
    float sacc[32];
    #pragma unroll
    for (int i = 0; i < 32; i++) sacc[i] = 0.0f;

    for (int p = 0; p < 3; p++) {
        __syncthreads();
        #pragma unroll
        for (int mt = 0; mt < 6; mt++) {
            #pragma unroll
            for (int ct = 0; ct < 2; ct++) {
                int col = wNb + ct * 16 + lr;
                #pragma unroll
                for (int q = 0; q < 8; q++) {
                    int row = wMb + mt * 16 + half * 8 + q;   // 0..191
                    if ((row >> 6) == p)
                        Ylds[(row & 63) * BCOLS + col] = acc[mt][ct][q] + biasLds[row];
                }
            }
        }
        __syncthreads();
        #pragma unroll
        for (int i = 0; i < 32; i++) {
            int idx = tid + i * 256;
            if (idx < CT * nl * 25) {
                int c   = idx / (nl * 25);
                int rem = idx - c * (nl * 25);
                int l = rem / 25, w = rem - (rem / 25) * 25;
                const float* yrow = &Ylds[c * BCOLS + l * 25];
                const float* acol = &Asc[p * 625 + w];
                float ps = 0.f;
                #pragma unroll
                for (int v = 0; v < 25; v++)
                    ps += yrow[v] * acol[v * 25];
                sacc[i] += ps;
            }
        }
    }
    #pragma unroll
    for (int i = 0; i < 32; i++) {
        int idx = tid + i * 256;
        if (idx < CT * nl * 25) {
            int c   = idx / (nl * 25);
            int rem = idx - c * (nl * 25);
            int l = rem / 25, w = rem - (rem / 25) * 25;
            s_out[(((size_t)n * 256 + c0 + c) * 1024 + l0 + l) * 25 + w] = sacc[i];
        }
    }
}

// ---------------------------------------------------------------------------
// Kernel 2: in-place causal sliding-window (length 9) sum along l.
// One thread owns one (n, c, v) column; register shift-window.
// ---------------------------------------------------------------------------
__global__ __launch_bounds__(256)
void stgcn_tempwin(float* __restrict__ s)
{
    int gid = blockIdx.x * 256 + threadIdx.x;
    if (gid >= 8 * 256 * 25) return;
    int n = gid / 6400;
    int r = gid - n * 6400;
    int c = r / 25, v = r - (r / 25) * 25;
    float* base = s + ((size_t)(n * 256 + c) * 1024) * 25 + v;

    float w[9];
    #pragma unroll
    for (int i = 0; i < 9; i++) w[i] = 0.0f;
    float sum = 0.0f;
    for (int l = 0; l < 1024; l++) {
        float val = base[(size_t)l * 25];
        sum += val - w[8];                 // w[8] holds val[l-9] (or 0)
        #pragma unroll
        for (int i = 8; i > 0; i--) w[i] = w[i - 1];
        w[0] = val;
        base[(size_t)l * 25] = sum;
    }
}

// ---------------------------------------------------------------------------
// Kernel 3: LayerNorm over (C,V) per (n,l) + affine + ReLU + residual + ReLU.
// ---------------------------------------------------------------------------
__global__ __launch_bounds__(256)
void stgcn_ln(const float* __restrict__ s,
              const float* __restrict__ x,
              const float* __restrict__ gamma,
              const float* __restrict__ beta,
              float* __restrict__ out)
{
    __shared__ float rs[256], rq[256];
    const int m = blockIdx.x, n = blockIdx.y, tid = threadIdx.x;

    float uv[25];
    float sum = 0.f, sq = 0.f;
    #pragma unroll
    for (int i = 0; i < 25; i++) {
        int idx = tid + i * 256;            // 0..6399
        int c = idx / 25, v = idx - c * 25;
        float u = s[(((size_t)n * 256 + c) * 1024 + m) * 25 + v];
        uv[i] = u;
        sum += u;
        sq  += u * u;
    }
    rs[tid] = sum; rq[tid] = sq;
    __syncthreads();
    for (int off = 128; off > 0; off >>= 1) {
        if (tid < off) { rs[tid] += rs[tid + off]; rq[tid] += rq[tid + off]; }
        __syncthreads();
    }
    const float mu   = rs[0] * (1.0f / 6400.0f);
    const float var  = rq[0] * (1.0f / 6400.0f) - mu * mu;
    const float rstd = rsqrtf(var + 1e-5f);

    #pragma unroll
    for (int i = 0; i < 25; i++) {
        int idx = tid + i * 256;
        int c = idx / 25, v = idx - c * 25;
        float h = (uv[i] - mu) * rstd * gamma[idx] + beta[idx];
        h = fmaxf(h, 0.0f);
        size_t xi = (((size_t)n * 256 + c) * 1024 + m) * 25 + v;
        out[xi] = fmaxf(h + x[xi], 0.0f);
    }
}

// ---------------------------------------------------------------------------
extern "C" void kernel_launch(void* const* d_in, const int* in_sizes, int n_in,
                              void* d_out, int out_size, void* d_ws, size_t ws_size,
                              hipStream_t stream)
{
    const float* x      = (const float*)d_in[0];
    const float* A1     = (const float*)d_in[1];
    const float* A2     = (const float*)d_in[2];
    const float* A3     = (const float*)d_in[3];
    const float* conv_w = (const float*)d_in[4];
    const float* conv_b = (const float*)d_in[5];
    const float* ln_g   = (const float*)d_in[6];
    const float* ln_b   = (const float*)d_in[7];
    const float* edge   = (const float*)d_in[8];
    float* out = (float*)d_out;
    float* s   = (float*)d_ws;   // N*Cout*L*V f32 = 209.7 MB scratch

    dim3 g1((1024 + LB - 1) / LB, 256 / CT, 8);   // (205, 4, 8)
    stgcn_gemm_spatial<<<g1, 256, 0, stream>>>(x, A1, A2, A3, conv_w, conv_b, edge, s);

    stgcn_tempwin<<<(8 * 256 * 25 + 255) / 256, 256, 0, stream>>>(s);

    dim3 g3(1024, 8);
    stgcn_ln<<<g3, 256, 0, stream>>>(s, x, ln_g, ln_b, out);
}
